// MultiHeadAttention_10299331576253
// MI455X (gfx1250) — matrix operand order repaired and checked by reading, not compiled, not run on hardware
//
#include <hip/hip_runtime.h>
#include <hip/hip_bf16.h>

// MI455X / gfx1250, wave32. Fused MHA block:
//   qkv projections (WMMA f16) -> flash attention (WMMA f16, online softmax)
//   -> out projection + residual (WMMA f16) -> LayerNorm.
// fp32 inputs converted to f16 on the fly; accumulation f32 via
// v_wmma_f32_16x16x32_f16. Workspace: 4 * (8192*1024) f16 = 64 MB.

typedef __attribute__((ext_vector_type(16))) _Float16 v16h;
typedef __attribute__((ext_vector_type(8)))  float    v8f;
typedef int v4i_raw __attribute__((vector_size(16)));

union FragA { uint4 q[2]; v16h h; };

__device__ __forceinline__ v8f wmma16(const v16h& a, const v16h& b, v8f c) {
  return __builtin_amdgcn_wmma_f32_16x16x32_f16(false, a, false, b, (short)0, c,
                                                false, false);
}

#define N_HEAD 16
#define D_HEAD 64
#define D_MODEL 1024

// Async global->LDS staging (ASYNCcnt path), guarded so compile never breaks.
// Probe-discovered signature: (v4i AS1* gaddr, v4i AS3* lds, imm off, imm cpol)
#if defined(__HIP_DEVICE_COMPILE__) && \
    __has_builtin(__builtin_amdgcn_global_load_async_to_lds_b128)
#define HAVE_ASYNC_LDS 1
__device__ __forceinline__ void async_copy16(const _Float16* g, _Float16* l) {
  __builtin_amdgcn_global_load_async_to_lds_b128(
      (__attribute__((address_space(1))) v4i_raw*)(g),
      (__attribute__((address_space(3))) v4i_raw*)(l), 0, 0);
}
#else
#define HAVE_ASYNC_LDS 0
#endif

__device__ __forceinline__ void wait_asynccnt0() {
#if defined(__HIP_DEVICE_COMPILE__) && \
    __has_builtin(__builtin_amdgcn_s_wait_asynccnt)
  __builtin_amdgcn_s_wait_asynccnt(0);
#elif defined(__HIP_DEVICE_COMPILE__)
  asm volatile("s_wait_asynccnt 0" ::: "memory");
#endif
}

// ---------------------------------------------------------------------------
// Tiled GEMM: C[M,N] = A[M,K] * W[K,N]; 256 threads = 8 waves, 128x64 tile,
// K-chunks of 32 staged through LDS as f16; each wave owns a 16x64 slice.
// MODE 0: write f16 head-major  [b,h,s,d]  (Q with alpha=1/8, K with alpha=1)
// MODE 1: write f16 d-major     [b,h,d,s]  (V transposed for PV fragment loads)
// MODE 2: A is f16 (ctx); write fp32 [m,n] + residual (output projection)
// ---------------------------------------------------------------------------
template <int MODE>
__global__ __launch_bounds__(256) void gemm_kernel(
    const float* __restrict__ Af, const _Float16* __restrict__ Ah,
    const float* __restrict__ W, _Float16* __restrict__ outH,
    float* __restrict__ outF, const float* __restrict__ residual,
    int M, int N, int K, int S, float alpha) {
  // Row stride 40 halves = 80 B (16B-multiple) so b128 LDS frag loads align.
  __shared__ _Float16 As[128][40];  // [m][k]  128x32 tile
  __shared__ _Float16 Bs[64][40];   // [n][k]   32x64 tile, stored transposed

  const int t = threadIdx.x;
  const int lane = t & 31, wave = t >> 5;
  const int m0 = lane & 15, hi = lane >> 4;
  const int row0 = blockIdx.y * 128;
  const int col0 = blockIdx.x * 64;

  v8f acc[4];
#pragma unroll
  for (int i = 0; i < 4; ++i) acc[i] = (v8f){0, 0, 0, 0, 0, 0, 0, 0};

  for (int k0 = 0; k0 < K; k0 += 32) {
    // Prefetch next K-chunk while we stage this one (global_prefetch_b8).
    if (k0 + 32 < K) {
      if constexpr (MODE == 2)
        __builtin_prefetch(&Ah[(size_t)(row0 + (t & 127)) * K + k0 + 32], 0, 1);
      else
        __builtin_prefetch(&Af[(size_t)(row0 + (t & 127)) * K + k0 + 32], 0, 1);
      __builtin_prefetch(
          &W[(size_t)(k0 + 32 + (t & 31)) * N + col0 + ((t >> 5) & 1) * 32], 0,
          1);
    }

    // ---- Stage A 128x32 into LDS (f16) ----
    if constexpr (MODE == 2) {
#if HAVE_ASYNC_LDS
      // A is already f16: raw async copy, 16 B per lane, 2 ops per thread.
#pragma unroll
      for (int i = 0; i < 2; ++i) {
        int e8 = i * 256 + t;  // 512 x 16B chunks
        int r = e8 >> 2, c0 = (e8 & 3) * 8;
        async_copy16(&Ah[(size_t)(row0 + r) * K + k0 + c0], &As[r][c0]);
      }
#else
#pragma unroll
      for (int i = 0; i < 16; ++i) {
        int e = i * 256 + t;
        int r = e >> 5, c = e & 31;
        As[r][c] = Ah[(size_t)(row0 + r) * K + k0 + c];
      }
#endif
    } else {
#pragma unroll
      for (int i = 0; i < 16; ++i) {
        int e = i * 256 + t;
        int r = e >> 5, c = e & 31;
        As[r][c] = (_Float16)Af[(size_t)(row0 + r) * K + k0 + c];
      }
    }

    // ---- Stage B 32x64 transposed into Bs[n][k] (fp32 -> f16) ----
#pragma unroll
    for (int i = 0; i < 8; ++i) {
      int e = i * 256 + t;
      int r = e >> 6, c = e & 63;  // r = k, c = n
      Bs[c][r] = (_Float16)W[(size_t)(k0 + r) * N + col0 + c];
    }

    if constexpr (MODE == 2) {
#if HAVE_ASYNC_LDS
      wait_asynccnt0();
#endif
    }
    __syncthreads();

    // A fragment: lane m0 holds row (wave*16+m0); half hi selects K offset +8.
    FragA a;
    const _Float16* ap = &As[wave * 16 + m0][0];
    a.q[0] = *(const uint4*)(ap + 8 * hi);       // K 0..7   (+8*hi)
    a.q[1] = *(const uint4*)(ap + 16 + 8 * hi);  // K 16..23 (+8*hi)

#pragma unroll
    for (int nf = 0; nf < 4; ++nf) {
      // B fragment: lane m0 = column; half hi selects K offset +8.
      FragA b;
      const _Float16* bp = &Bs[nf * 16 + m0][0];
      b.q[0] = *(const uint4*)(bp + 8 * hi);       // K 0..7   (+8*hi)
      b.q[1] = *(const uint4*)(bp + 16 + 8 * hi);  // K 16..23 (+8*hi)
      acc[nf] = wmma16(a.h, b.h, acc[nf]);
    }
    __syncthreads();
  }

  // Epilogue. C layout: vgpr r -> M = r + 8*hi, N across 16 lanes.
#pragma unroll
  for (int nf = 0; nf < 4; ++nf) {
#pragma unroll
    for (int r = 0; r < 8; ++r) {
      float val = acc[nf][r] * alpha;
      int gm = row0 + wave * 16 + r + 8 * hi;
      int gn = col0 + nf * 16 + m0;
      if constexpr (MODE == 2) {
        size_t idx = (size_t)gm * N + gn;
        outF[idx] = val + residual[idx];
      } else {
        int bI = gm / S, sI = gm % S;
        int hI = gn >> 6, dI = gn & 63;
        if constexpr (MODE == 0)
          outH[(((size_t)bI * N_HEAD + hI) * S + sI) * D_HEAD + dI] =
              (_Float16)val;
        else
          outH[(((size_t)bI * N_HEAD + hI) * D_HEAD + dI) * S + sI] =
              (_Float16)val;
      }
    }
  }
}

// ---------------------------------------------------------------------------
// Flash attention: per (b,h, 64-query tile) block; each wave owns 16 rows.
// scores = (Q*alpha) K^T (scale folded into Q proj), online softmax, ctx += PV.
// qh: [bh][s][64] f16, kh: [bh][s][64] f16, vT: [bh][d][S] f16
// ctx out: [b*s][1024] f16 (head-concatenated, ready for the FC GEMM)
// ---------------------------------------------------------------------------
__global__ __launch_bounds__(128) void attn_kernel(
    const _Float16* __restrict__ qh, const _Float16* __restrict__ kh,
    const _Float16* __restrict__ vT, _Float16* __restrict__ ctxb, int S) {
  __shared__ _Float16 P[4][16][32];  // per-wave probability tile

  const int nt = S >> 6;
  const int bh = blockIdx.x / nt;
  const int tile = blockIdx.x % nt;
  const int b = bh / N_HEAD, h = bh % N_HEAD;
  const int lane = threadIdx.x & 31, wave = threadIdx.x >> 5;
  const int m0 = lane & 15, hi = lane >> 4;
  const int q0 = tile * 64 + wave * 16;

  // Q A-fragments for d 0..31 and d 32..63 (loaded once).
  const _Float16* qp = qh + ((size_t)bh * S + q0 + m0) * D_HEAD;
  FragA qa0, qa1;
  qa0.q[0] = *(const uint4*)(qp + 8 * hi);
  qa0.q[1] = *(const uint4*)(qp + 16 + 8 * hi);
  qa1.q[0] = *(const uint4*)(qp + 32 + 8 * hi);
  qa1.q[1] = *(const uint4*)(qp + 48 + 8 * hi);

  v8f acc[4];
#pragma unroll
  for (int i = 0; i < 4; ++i) acc[i] = (v8f){0, 0, 0, 0, 0, 0, 0, 0};
  float mrun[8], lrun[8];
#pragma unroll
  for (int r = 0; r < 8; ++r) { mrun[r] = -1e30f; lrun[r] = 0.0f; }

  for (int key0 = 0; key0 < S; key0 += 32) {
    // ---- scores: two 16x16 tiles (keys key0..+15, key0+16..+31) ----
    v8f s0 = (v8f){0, 0, 0, 0, 0, 0, 0, 0};
    v8f s1 = (v8f){0, 0, 0, 0, 0, 0, 0, 0};
    const _Float16* kp0 = kh + ((size_t)bh * S + key0 + m0) * D_HEAD;
    const _Float16* kp1 = kp0 + 16 * D_HEAD;
    FragA kb;
    kb.q[0] = *(const uint4*)(kp0 + 8 * hi);
    kb.q[1] = *(const uint4*)(kp0 + 16 + 8 * hi);
    s0 = wmma16(qa0.h, kb.h, s0);
    kb.q[0] = *(const uint4*)(kp0 + 32 + 8 * hi);
    kb.q[1] = *(const uint4*)(kp0 + 48 + 8 * hi);
    s0 = wmma16(qa1.h, kb.h, s0);
    kb.q[0] = *(const uint4*)(kp1 + 8 * hi);
    kb.q[1] = *(const uint4*)(kp1 + 16 + 8 * hi);
    s1 = wmma16(qa0.h, kb.h, s1);
    kb.q[0] = *(const uint4*)(kp1 + 32 + 8 * hi);
    kb.q[1] = *(const uint4*)(kp1 + 48 + 8 * hi);
    s1 = wmma16(qa1.h, kb.h, s1);

    // ---- online softmax (row M=r+8*hi lives in one 16-lane half) ----
    float p0[8], p1[8], corr[8];
#pragma unroll
    for (int r = 0; r < 8; ++r) {
      float mx = fmaxf(s0[r], s1[r]);
      mx = fmaxf(mx, __shfl_xor(mx, 1, 16));
      mx = fmaxf(mx, __shfl_xor(mx, 2, 16));
      mx = fmaxf(mx, __shfl_xor(mx, 4, 16));
      mx = fmaxf(mx, __shfl_xor(mx, 8, 16));
      float mnew = fmaxf(mrun[r], mx);
      corr[r] = __expf(mrun[r] - mnew);
      p0[r] = __expf(s0[r] - mnew);
      p1[r] = __expf(s1[r] - mnew);
      float rs = p0[r] + p1[r];
      rs += __shfl_xor(rs, 1, 16);
      rs += __shfl_xor(rs, 2, 16);
      rs += __shfl_xor(rs, 4, 16);
      rs += __shfl_xor(rs, 8, 16);
      lrun[r] = lrun[r] * corr[r] + rs;
      mrun[r] = mnew;
    }
#pragma unroll
    for (int nf = 0; nf < 4; ++nf)
#pragma unroll
      for (int r = 0; r < 8; ++r) acc[nf][r] *= corr[r];

    // ---- re-layout P (C layout) -> A fragment via per-wave LDS ----
#pragma unroll
    for (int r = 0; r < 8; ++r) {
      int row = r + 8 * hi;
      P[wave][row][m0] = (_Float16)p0[r];
      P[wave][row][16 + m0] = (_Float16)p1[r];
    }
    // Same-wave LDS ops are in-order; just drain DScnt before reading back.
    asm volatile("s_wait_dscnt 0" ::: "memory");
    FragA pa;
    const _Float16* pp = &P[wave][m0][0];
    pa.q[0] = *(const uint4*)(pp + 8 * hi);
    pa.q[1] = *(const uint4*)(pp + 16 + 8 * hi);

    // ---- ctx += P (16x32) * V (32x64), V stored d-major for contiguous K ----
    const _Float16* vbase = vT + (size_t)bh * D_HEAD * S + key0;
#pragma unroll
    for (int nf = 0; nf < 4; ++nf) {
      const _Float16* vp = vbase + (size_t)(nf * 16 + m0) * S;
      FragA vb;
      vb.q[0] = *(const uint4*)(vp + 8 * hi);
      vb.q[1] = *(const uint4*)(vp + 16 + 8 * hi);
      acc[nf] = wmma16(pa.h, vb.h, acc[nf]);
    }
  }

  // ---- normalize and store ctx [b*s][h*64+d] as f16 ----
#pragma unroll
  for (int nf = 0; nf < 4; ++nf)
#pragma unroll
    for (int r = 0; r < 8; ++r) {
      int gr = b * S + q0 + r + 8 * hi;
      float v = acc[nf][r] / lrun[r];
      ctxb[(size_t)gr * D_MODEL + h * D_HEAD + nf * 16 + m0] = (_Float16)v;
    }
}

// ---------------------------------------------------------------------------
// LayerNorm over last dim (1024), one 256-thread block per row, in-place.
// ---------------------------------------------------------------------------
__global__ __launch_bounds__(256) void ln_kernel(float* __restrict__ out,
                                                 const float* __restrict__ gamma,
                                                 const float* __restrict__ beta,
                                                 int N) {
  __shared__ float wsum[8], wsq[8];
  const int lane = threadIdx.x & 31, wave = threadIdx.x >> 5;
  float* p = out + (size_t)blockIdx.x * N;
  float s = 0.0f, ss = 0.0f;
  for (int i = threadIdx.x; i < N; i += 256) {
    float x = p[i];
    s += x;
    ss += x * x;
  }
#pragma unroll
  for (int off = 16; off >= 1; off >>= 1) {
    s += __shfl_xor(s, off, 32);
    ss += __shfl_xor(ss, off, 32);
  }
  if (lane == 0) { wsum[wave] = s; wsq[wave] = ss; }
  __syncthreads();
  float ts = 0.0f, tss = 0.0f;
#pragma unroll
  for (int w = 0; w < 8; ++w) { ts += wsum[w]; tss += wsq[w]; }
  float mean = ts / N;
  float var = tss / N - mean * mean;
  float rstd = rsqrtf(var + 1e-6f);
  for (int i = threadIdx.x; i < N; i += 256) {
    float x = p[i];
    p[i] = (x - mean) * rstd * gamma[i] + beta[i];
  }
}

// ---------------------------------------------------------------------------
extern "C" void kernel_launch(void* const* d_in, const int* in_sizes, int n_in,
                              void* d_out, int out_size, void* d_ws,
                              size_t ws_size, hipStream_t stream) {
  const float* q = (const float*)d_in[0];
  const float* k = (const float*)d_in[1];
  const float* v = (const float*)d_in[2];
  const float* w_q = (const float*)d_in[3];
  const float* w_k = (const float*)d_in[4];
  const float* w_v = (const float*)d_in[5];
  const float* w_fc = (const float*)d_in[6];
  const float* gamma = (const float*)d_in[7];
  const float* beta = (const float*)d_in[8];

  const int M = in_sizes[0] / D_MODEL;  // B*S = 8192
  const int B = 4;
  const int S = M / B;  // 2048

  // Workspace: 4 f16 buffers of M*1024 elements each (qh, kh, vT, ctx) = 64 MB.
  size_t szH = (size_t)M * D_MODEL;
  _Float16* qhW = (_Float16*)d_ws;
  _Float16* khW = qhW + szH;
  _Float16* vTW = khW + szH;
  _Float16* ctxW = vTW + szH;

  dim3 gg(D_MODEL / 64, M / 128);  // (16, 64)
  const float qscale = 0.125f;     // 1/sqrt(D_HEAD), folded into Q projection

  gemm_kernel<0><<<gg, 256, 0, stream>>>(q, nullptr, w_q, qhW, nullptr, nullptr,
                                         M, D_MODEL, D_MODEL, S, qscale);
  gemm_kernel<0><<<gg, 256, 0, stream>>>(k, nullptr, w_k, khW, nullptr, nullptr,
                                         M, D_MODEL, D_MODEL, S, 1.0f);
  gemm_kernel<1><<<gg, 256, 0, stream>>>(v, nullptr, w_v, vTW, nullptr, nullptr,
                                         M, D_MODEL, D_MODEL, S, 1.0f);

  int attn_blocks = B * N_HEAD * (S / 64);  // 2048
  attn_kernel<<<attn_blocks, 128, 0, stream>>>(qhW, khW, vTW, ctxW, S);

  gemm_kernel<2><<<gg, 256, 0, stream>>>(nullptr, ctxW, w_fc, nullptr,
                                         (float*)d_out, q, M, D_MODEL, D_MODEL,
                                         S, 1.0f);

  ln_kernel<<<M, 256, 0, stream>>>((float*)d_out, gamma, beta, D_MODEL);
}
